// Encoder_75256416960604
// MI455X (gfx1250) — matrix operand order repaired and checked by reading, not compiled, not run on hardware
//
#include <hip/hip_runtime.h>

typedef __attribute__((ext_vector_type(16))) __bf16 v16bf;
typedef __attribute__((ext_vector_type(8)))  float  v8f;
typedef __attribute__((ext_vector_type(4)))  unsigned int v4u;

#define TT 64      // sequence length
#define EE 256     // embed dim
#define DD 64      // head dim
#define KTOT 16384 // T*E
#define NOUT 256
#define SPB 72     // bf16 LDS row stride (144 B = 9*16 B -> every frag 16B-aligned)
#define SPF 68     // f32 LDS row stride (scores)

union FragBf { v16bf v; v4u u[2]; };

// A-matrix 16x32 bf16 fragment: plane layout [row][k]; base = &plane[m*SPB + kb]
// lane L: VGPR0-3 = A[m][kb + 8*half .. +7], VGPR4-7 = A[m][kb+16+8*half .. +7]
__device__ __forceinline__ v16bf ld_afrag(const __bf16* base, int half) {
    FragBf f;
    f.u[0] = *(const v4u*)(base + 8 * half);
    f.u[1] = *(const v4u*)(base + 16 + 8 * half);
    return f.v;
}
// B-matrix 32x16 bf16 fragment: plane layout [col][k]; base = &plane[n*SPB + kb]
// lane L: VGPR0-3 = B[kb + 8*half .. +7][n], VGPR4-7 = B[kb+16+8*half .. +7][n]
__device__ __forceinline__ v16bf ld_bfrag(const __bf16* base, int half) {
    FragBf f;
    f.u[0] = *(const v4u*)(base + 8 * half);
    f.u[1] = *(const v4u*)(base + 16 + 8 * half);
    return f.v;
}
__device__ __forceinline__ unsigned pack2(__bf16 a, __bf16 b) {
    unsigned short ua = __builtin_bit_cast(unsigned short, a);
    unsigned short ub = __builtin_bit_cast(unsigned short, b);
    return (unsigned)ua | ((unsigned)ub << 16);
}

#define WMMA_BF16(A, B, C) \
    __builtin_amdgcn_wmma_f32_16x16x32_bf16(false, (A), false, (B), (short)0, (C), false, false)

// Per-row sparse softmax: allowed keys j = {r, r+8, ..., i}, r = i%8. Reads only.
__device__ __forceinline__ void softmax_row(const float* ss, int i, float* pv) {
    const int r = i & 7, nal = (i >> 3) + 1;
    float mx = -1e30f;
#pragma unroll
    for (int k = 0; k < 8; ++k) {
        float s = (k < nal) ? ss[i * SPF + r + 8 * k] : -1e30f;
        mx = fmaxf(mx, s);
    }
    float z = 0.f;
#pragma unroll
    for (int k = 0; k < 8; ++k) {
        float s = (k < nal) ? ss[i * SPF + r + 8 * k] : -1e30f;
        float e = __expf(s - mx);
        e = (k < nal) ? e : 0.f;
        pv[k] = e;
        z += e;
    }
    const float inv = 1.f / z;
#pragma unroll
    for (int k = 0; k < 8; ++k) pv[k] *= inv;
}

// Write one probability row as hi/lo bf16 planes (zeros on disallowed cols).
__device__ __forceinline__ void write_prow(__bf16* p_hi, __bf16* p_lo, int i, const float* pv) {
    const int r = i & 7;
#pragma unroll
    for (int j2 = 0; j2 < 32; ++j2) {
        const int j0 = 2 * j2, j1 = j0 + 1;
        float v0 = ((j0 & 7) == r) ? pv[j0 >> 3] : 0.f;
        float v1 = ((j1 & 7) == r) ? pv[j1 >> 3] : 0.f;
        __bf16 h0 = (__bf16)v0, h1 = (__bf16)v1;
        __bf16 l0 = (__bf16)(v0 - (float)h0), l1 = (__bf16)(v1 - (float)h1);
        *(unsigned*)(p_hi + i * SPB + j0) = pack2(h0, h1);
        *(unsigned*)(p_lo + i * SPB + j0) = pack2(l0, l1);
    }
}

// ---------------- Kernel 1: strided attention, one wave per (b, head) -------
__global__ __launch_bounds__(32)
void attn_strided_kernel(const float* __restrict__ x, float* __restrict__ attn) {
    // 54 KB: qk hi/lo [t][c], vT hi/lo [c][t], {scores f32 | P hi/lo} (aliased)
    __shared__ __attribute__((aligned(16))) char smem[55296];
    __bf16* qk_hi = (__bf16*)(smem);
    __bf16* qk_lo = (__bf16*)(smem + 9216);
    __bf16* vt_hi = (__bf16*)(smem + 18432);
    __bf16* vt_lo = (__bf16*)(smem + 27648);
    float*  ss    = (float*) (smem + 36864);
    __bf16* p_hi  = (__bf16*)(smem + 36864);   // aliases ss (barriered)
    __bf16* p_lo  = (__bf16*)(smem + 46080);

    const int bh   = blockIdx.x;
    const int b    = bh >> 2;
    const int hd   = bh & 3;
    const int lane = threadIdx.x;   // wave32
    const float* xb = x + (size_t)b * TT * EE + hd * DD;

    // Stage 64x64 head slice, split fp32 -> bf16 hi + lo residual, both
    // row-major ([t][c], Q and K^T reads) and transposed ([c][t], V reads).
    for (int idx = lane; idx < TT * (DD / 4); idx += 32) {
        const int t  = idx >> 4;
        const int c4 = (idx & 15) * 4;
        const float4 v = *(const float4*)(xb + t * EE + c4);
        float f[4] = {v.x, v.y, v.z, v.w};
        __bf16 hh[4], ll[4];
#pragma unroll
        for (int j = 0; j < 4; ++j) {
            hh[j] = (__bf16)f[j];
            ll[j] = (__bf16)(f[j] - (float)hh[j]);
        }
        uint2 ph; ph.x = pack2(hh[0], hh[1]); ph.y = pack2(hh[2], hh[3]);
        uint2 pl; pl.x = pack2(ll[0], ll[1]); pl.y = pack2(ll[2], ll[3]);
        *(uint2*)(qk_hi + t * SPB + c4) = ph;
        *(uint2*)(qk_lo + t * SPB + c4) = pl;
#pragma unroll
        for (int j = 0; j < 4; ++j) {
            vt_hi[(c4 + j) * SPB + t] = hh[j];
            vt_lo[(c4 + j) * SPB + t] = ll[j];
        }
    }
    __syncthreads();

    const int mrow = lane & 15;
    const int half = lane >> 4;
    const int drow = half << 3;

    // S = Q K^T * (1/sqrt(64)), bf16x3: hi*hi + hi*lo + lo*hi
    for (int mi = 0; mi < 4; ++mi) {
        v16bf ah[2], al[2];
#pragma unroll
        for (int kb = 0; kb < 2; ++kb) {
            ah[kb] = ld_afrag(qk_hi + (mi * 16 + mrow) * SPB + kb * 32, half);
            al[kb] = ld_afrag(qk_lo + (mi * 16 + mrow) * SPB + kb * 32, half);
        }
        for (int ni = 0; ni < 4; ++ni) {
            v8f acc = {};
#pragma unroll
            for (int kb = 0; kb < 2; ++kb) {
                v16bf bhh = ld_bfrag(qk_hi + (ni * 16 + mrow) * SPB + kb * 32, half);
                v16bf bll = ld_bfrag(qk_lo + (ni * 16 + mrow) * SPB + kb * 32, half);
                acc = WMMA_BF16(ah[kb], bhh, acc);
                acc = WMMA_BF16(ah[kb], bll, acc);
                acc = WMMA_BF16(al[kb], bhh, acc);
            }
#pragma unroll
            for (int v = 0; v < 8; ++v)
                ss[(mi * 16 + v + drow) * SPF + ni * 16 + mrow] = acc[v] * 0.125f;
        }
    }
    __syncthreads();

    // Softmax: all reads of ss into registers, then write P planes (alias ss).
    float pv0[8], pv1[8];
    softmax_row(ss, lane, pv0);
    softmax_row(ss, lane + 32, pv1);
    __syncthreads();
    write_prow(p_hi, p_lo, lane, pv0);
    write_prow(p_hi, p_lo, lane + 32, pv1);
    __syncthreads();

    // A_out = P V, bf16x3; B[k][n] = V[k][n] -> vt plane [n][k]
    float* ap = attn + (size_t)b * (TT * EE) + hd * DD;
    for (int mi = 0; mi < 4; ++mi) {
        v16bf ah2[2], al2[2];
#pragma unroll
        for (int kb = 0; kb < 2; ++kb) {
            ah2[kb] = ld_afrag(p_hi + (mi * 16 + mrow) * SPB + kb * 32, half);
            al2[kb] = ld_afrag(p_lo + (mi * 16 + mrow) * SPB + kb * 32, half);
        }
        for (int ni = 0; ni < 4; ++ni) {
            v8f acc = {};
#pragma unroll
            for (int kb = 0; kb < 2; ++kb) {
                v16bf bhh = ld_bfrag(vt_hi + (ni * 16 + mrow) * SPB + kb * 32, half);
                v16bf bll = ld_bfrag(vt_lo + (ni * 16 + mrow) * SPB + kb * 32, half);
                acc = WMMA_BF16(ah2[kb], bhh, acc);
                acc = WMMA_BF16(ah2[kb], bll, acc);
                acc = WMMA_BF16(al2[kb], bhh, acc);
            }
#pragma unroll
            for (int v = 0; v < 8; ++v)
                ap[(mi * 16 + v + drow) * EE + ni * 16 + mrow] = acc[v];
        }
    }
}

// ---------------- Kernel 2: FFN  relu((attn+x) @ W1 + b1) --------------------
// M=4096, K=16384, N=256. 64x64 tile / block, 8 waves (4 along M x 2 along N).
__global__ __launch_bounds__(256)
void ffn_kernel(const float* __restrict__ x, const float* __restrict__ attn,
                const float* __restrict__ W1, const float* __restrict__ bias,
                float* __restrict__ out) {
    __shared__ __attribute__((aligned(16))) char smem[36864];  // 36 KB
    __bf16* ha_hi = (__bf16*)(smem);            // h tile [m][k]
    __bf16* ha_lo = (__bf16*)(smem + 9216);
    __bf16* wt_hi = (__bf16*)(smem + 18432);    // W1 tile transposed [n][k]
    __bf16* wt_lo = (__bf16*)(smem + 27648);

    const int m0   = (blockIdx.x >> 2) * 64;
    const int n0   = (blockIdx.x & 3) * 64;
    const int tid  = threadIdx.x;
    const int lane = tid & 31;
    const int wave = tid >> 5;
    const int wm   = wave & 3;
    const int wn   = wave >> 2;
    const int mrow = lane & 15;
    const int half = lane >> 4;
    const int drow = half << 3;

    v8f acc0 = {}, acc1 = {};

    for (int kc = 0; kc < KTOT; kc += 64) {
        // h tile = attn + x, fused on load; split hi/lo, pack pairs along k
        for (int idx = tid; idx < 64 * 16; idx += 256) {
            const int r  = idx >> 4;
            const int c4 = (idx & 15) * 4;
            const size_t g = (size_t)(m0 + r) * KTOT + kc + c4;
            const float4 va = *(const float4*)(attn + g);
            const float4 vx = *(const float4*)(x + g);
            float f[4] = {va.x + vx.x, va.y + vx.y, va.z + vx.z, va.w + vx.w};
            __bf16 hh[4], ll[4];
#pragma unroll
            for (int j = 0; j < 4; ++j) {
                hh[j] = (__bf16)f[j];
                ll[j] = (__bf16)(f[j] - (float)hh[j]);
            }
            uint2 ph; ph.x = pack2(hh[0], hh[1]); ph.y = pack2(hh[2], hh[3]);
            uint2 pl; pl.x = pack2(ll[0], ll[1]); pl.y = pack2(ll[2], ll[3]);
            *(uint2*)(ha_hi + r * SPB + c4) = ph;
            *(uint2*)(ha_lo + r * SPB + c4) = pl;
        }
        // W1 tile: coalesced float4 along n, transposed bf16 stores -> [n][k]
        for (int idx = tid; idx < 64 * 16; idx += 256) {
            const int r  = idx >> 4;           // k
            const int c4 = (idx & 15) * 4;     // n offset
            const float4 vw = *(const float4*)(W1 + (size_t)(kc + r) * NOUT + n0 + c4);
            float f[4] = {vw.x, vw.y, vw.z, vw.w};
#pragma unroll
            for (int j = 0; j < 4; ++j) {
                __bf16 hv = (__bf16)f[j];
                wt_hi[(c4 + j) * SPB + r] = hv;
                wt_lo[(c4 + j) * SPB + r] = (__bf16)(f[j] - (float)hv);
            }
        }
        __syncthreads();

        v16bf ah[2], al[2];
#pragma unroll
        for (int kb = 0; kb < 2; ++kb) {
            ah[kb] = ld_afrag(ha_hi + (wm * 16 + mrow) * SPB + kb * 32, half);
            al[kb] = ld_afrag(ha_lo + (wm * 16 + mrow) * SPB + kb * 32, half);
        }
        const int nc0 = wn * 32 + mrow;
#pragma unroll
        for (int kb = 0; kb < 2; ++kb) {
            v16bf bh0 = ld_bfrag(wt_hi + nc0 * SPB + kb * 32, half);
            v16bf bl0 = ld_bfrag(wt_lo + nc0 * SPB + kb * 32, half);
            acc0 = WMMA_BF16(ah[kb], bh0, acc0);
            acc0 = WMMA_BF16(ah[kb], bl0, acc0);
            acc0 = WMMA_BF16(al[kb], bh0, acc0);
            v16bf bh1 = ld_bfrag(wt_hi + (nc0 + 16) * SPB + kb * 32, half);
            v16bf bl1 = ld_bfrag(wt_lo + (nc0 + 16) * SPB + kb * 32, half);
            acc1 = WMMA_BF16(ah[kb], bh1, acc1);
            acc1 = WMMA_BF16(ah[kb], bl1, acc1);
            acc1 = WMMA_BF16(al[kb], bh1, acc1);
        }
        __syncthreads();
    }

    // bias + relu + store
    const int col0 = n0 + wn * 32 + mrow;
    const float b0 = bias[col0];
    const float b1 = bias[col0 + 16];
#pragma unroll
    for (int v = 0; v < 8; ++v) {
        const int row = m0 + wm * 16 + v + drow;
        float r0 = acc0[v] + b0;
        float r1 = acc1[v] + b1;
        out[(size_t)row * NOUT + col0]      = r0 > 0.f ? r0 : 0.f;
        out[(size_t)row * NOUT + col0 + 16] = r1 > 0.f ? r1 : 0.f;
    }
}

extern "C" void kernel_launch(void* const* d_in, const int* in_sizes, int n_in,
                              void* d_out, int out_size, void* d_ws, size_t ws_size,
                              hipStream_t stream) {
    (void)in_sizes; (void)n_in; (void)out_size; (void)d_ws; (void)ws_size;
    const float* x  = (const float*)d_in[0];   // [4096,64,256]
    // d_in[1] = mask (unused)
    const float* W  = (const float*)d_in[2];   // [2,16384,256]
    const float* bb = (const float*)d_in[3];   // [2,256]

    float* out  = (float*)d_out;                          // [4096,256]
    float* attn = out + (size_t)4096 * 256;               // [4096,64,256]
    const float* W1 = W  + (size_t)16384 * 256;           // W[-1]
    const float* b1 = bb + 256;                           // b[-1]

    attn_strided_kernel<<<4096 * 4, 32, 0, stream>>>(x, attn);
    ffn_kernel<<<64 * 4, 256, 0, stream>>>(x, attn, W1, b1, out);
}
